// GaugeCovariantAttention_12661563589246
// MI455X (gfx1250) — matrix operand order repaired and checked by reading, not compiled, not run on hardware
//
#include <hip/hip_runtime.h>
#include <cstddef>

// ---------------------------------------------------------------------------
// GaugeCovariantAttention for MI455X (gfx1250), wave32 + WMMA bf16.
//
// Pipeline (all on `stream`):
//  0. zero attn_avg accumulator
//  1. hpre = gelu(x@W1+b1)            [WMMA GEMM, f32 out]
//  2. hln  = LayerNorm(hpre)
//  3. Abf  = bf16(hln@W2+b2)          [WMMA GEMM, 137 GFLOP, bf16 out]
//  4. Q,K,V = x@W{q,k,v}+b            [WMMA GEMMs]
//  5. softmax attention -> xattn, attn_avg (+= p/H via f32 atomics)
//  6. x_out = x + xattn@Wo + bo       [WMMA GEMM, residual, -> d_out]
//  7. lie_j[b,j,mn] = sum_d x*A
//  8. uch = u@Wfp+bfp                 [WMMA GEMM]
//  9. fused: lie_i tile via WMMA, Gamma = exp(-lie) Taylor-6 in registers,
//     u_t += attn_avg * Gamma @ u_chunks  (no 134MB lie/Gamma intermediates)
// 10. u_out = u + u_t@Wfo + bfo       [WMMA GEMM, residual, -> d_out+524288]
// ---------------------------------------------------------------------------

typedef __attribute__((ext_vector_type(16))) __bf16 v16bf;
typedef __attribute__((ext_vector_type(8)))  __bf16 v8bf;
typedef __attribute__((ext_vector_type(8)))  float  v8f;

#define BB 2
#define SS 512
#define DD 512
#define HH 8
#define DH 64
#define KK 8

// ---------------------------------------------------------------------------
// Generic bf16-WMMA GEMM: C[M,N] = act(A_f32[M,K] @ B_f32[K,N] + bias)
// (+ optional f32 residual), writes f32 and/or bf16.  Tile 64x64, K-step 32,
// 128 threads = 4 waves, each wave owns a 16-row stripe (4 wmma per K-step).
// f32 -> bf16 conversion happens while staging tiles into LDS.
// ---------------------------------------------------------------------------
__global__ __launch_bounds__(128) void wmma_gemm_kernel(
    const float* __restrict__ A, const float* __restrict__ B,
    const float* __restrict__ bias, const float* __restrict__ residual,
    float* __restrict__ outF, __bf16* __restrict__ outBF,
    int M, int N, int K, int act)
{
    __shared__ __align__(32) __bf16 Alds[64 * 32];
    __shared__ __align__(32) __bf16 Blds[64 * 32];

    const int n0 = blockIdx.x * 64;
    const int m0 = blockIdx.y * 64;
    const int t = threadIdx.x;
    const int wave = t >> 5;
    const int lane = t & 31;
    const int lLo = lane & 15;
    const int lHi = lane >> 4;

    v8f cacc[4];
    #pragma unroll
    for (int tn = 0; tn < 4; ++tn) cacc[tn] = (v8f){};

    for (int kk = 0; kk < K; kk += 32) {
        __syncthreads();
        // stage A tile 64x32 (row-major), converting f32->bf16
        {
            int row = t >> 1, seg = (t & 1) * 16;
            const float* src = A + (size_t)(m0 + row) * K + kk + seg;
            #pragma unroll
            for (int e = 0; e < 16; ++e)
                Alds[row * 32 + seg + e] = (__bf16)src[e];
        }
        // stage B tile 32x64 transposed into Blds[n][k]
        {
            int k = t >> 2, nseg = (t & 3) * 16;
            const float* src = B + (size_t)(kk + k) * N + n0 + nseg;
            #pragma unroll
            for (int e = 0; e < 16; ++e)
                Blds[(nseg + e) * 32 + k] = (__bf16)src[e];
        }
        __syncthreads();

        // A operand: lane holds M=lLo, K = lHi*16..+15
        const int arow = wave * 16 + lLo;
        union { v16bf v; v8bf h[2]; } ua;
        ua.h[0] = *(const v8bf*)&Alds[arow * 32 + lHi * 16];
        ua.h[1] = *(const v8bf*)&Alds[arow * 32 + lHi * 16 + 8];

        #pragma unroll
        for (int tn = 0; tn < 4; ++tn) {
            // B operand: lane holds N=lLo (of subtile tn), K = lHi*16..+15
            v16bf bm = *(const v16bf*)&Blds[(tn * 16 + lLo) * 32 + lHi * 16];
            cacc[tn] = __builtin_amdgcn_wmma_f32_16x16x32_bf16(
                false, ua.v, false, bm, (short)0, cacc[tn], false, false);
        }
    }

    // epilogue: C layout lane(0-15): M=r, lane(16-31): M=8+r ; N=lLo
    #pragma unroll
    for (int tn = 0; tn < 4; ++tn) {
        #pragma unroll
        for (int r = 0; r < 8; ++r) {
            int row = m0 + wave * 16 + lHi * 8 + r;
            int col = n0 + tn * 16 + lLo;
            float v = cacc[tn][r] + bias[col];
            if (act == 1)  // exact GELU
                v = 0.5f * v * (1.0f + erff(v * 0.70710678118654752f));
            size_t idx = (size_t)row * N + col;
            if (residual) v += residual[idx];
            if (outF)  outF[idx] = v;
            if (outBF) outBF[idx] = (__bf16)v;
        }
    }
}

// ---------------------------------------------------------------------------
__global__ __launch_bounds__(256) void zero_kernel(float* __restrict__ p, int n)
{
    int i = blockIdx.x * blockDim.x + threadIdx.x;
    if (i < n) p[i] = 0.0f;
}

// LayerNorm over last dim (2048), one row per block.
__global__ __launch_bounds__(256) void ln_kernel(
    const float* __restrict__ h, const float* __restrict__ g,
    const float* __restrict__ bb, float* __restrict__ out, int N)
{
    __shared__ float rs[256], rq[256];
    const int row = blockIdx.x, t = threadIdx.x;
    const float* src = h + (size_t)row * N;
    float s = 0.f, s2 = 0.f;
    for (int c = t; c < N; c += 256) { float v = src[c]; s += v; s2 += v * v; }
    rs[t] = s; rq[t] = s2; __syncthreads();
    for (int off = 128; off > 0; off >>= 1) {
        if (t < off) { rs[t] += rs[t + off]; rq[t] += rq[t + off]; }
        __syncthreads();
    }
    float mu = rs[0] / N;
    float var = rq[0] / N - mu * mu;
    float rstd = rsqrtf(var + 1e-5f);
    float* dst = out + (size_t)row * N;
    for (int c = t; c < N; c += 256)
        dst[c] = (src[c] - mu) * rstd * g[c] + bb[c];
}

// lie_j[b,j,mn] = sum_d x[b,j,d] * A[b,j,d,mn]   (64 threads = mn)
__global__ __launch_bounds__(64) void liej_kernel(
    const float* __restrict__ x, const __bf16* __restrict__ Abf,
    float* __restrict__ liej)
{
    const int row = blockIdx.x;   // b*S + j
    const int mn = threadIdx.x;
    const float* xr = x + (size_t)row * DD;
    const __bf16* Ar = Abf + (size_t)row * DD * 64;
    float s = 0.f;
    for (int d = 0; d < DD; ++d)
        s += xr[d] * (float)Ar[(size_t)d * 64 + mn];
    liej[(size_t)row * 64 + mn] = s;
}

// ---------------------------------------------------------------------------
// Attention: one block per (b,h,i). scores -> softmax -> xattn;
// accumulates attn_avg[b,i,j] += p/H with f32 global atomics.
// ---------------------------------------------------------------------------
__global__ __launch_bounds__(128) void attn_kernel(
    const float* __restrict__ Qf, const float* __restrict__ Kf,
    const float* __restrict__ Vf, float* __restrict__ attnavg,
    float* __restrict__ xattn)
{
    const int i = blockIdx.x, h = blockIdx.y, b = blockIdx.z;
    const int t = threadIdx.x;
    __shared__ float qld[64];
    __shared__ float sc[SS];
    __shared__ float red[128];

    if (t < 64) qld[t] = Qf[((size_t)(b * SS + i)) * DD + h * 64 + t];
    __syncthreads();

    float loc[4];
    #pragma unroll
    for (int q = 0; q < 4; ++q) {
        int j = q * 128 + t;
        const float* kr = Kf + ((size_t)(b * SS + j)) * DD + h * 64;
        float dot = 0.f;
        for (int c = 0; c < 64; ++c) dot += kr[c] * qld[c];
        loc[q] = dot * 0.125f;            // / sqrt(dh)=8
        sc[j] = loc[q];
    }
    // row max
    float m = fmaxf(fmaxf(loc[0], loc[1]), fmaxf(loc[2], loc[3]));
    red[t] = m; __syncthreads();
    for (int off = 64; off > 0; off >>= 1) {
        if (t < off) red[t] = fmaxf(red[t], red[t + off]);
        __syncthreads();
    }
    float mx = red[0]; __syncthreads();
    // exp + sum
    float ps = 0.f;
    #pragma unroll
    for (int q = 0; q < 4; ++q) {
        int j = q * 128 + t;
        float e = __expf(loc[q] - mx);
        sc[j] = e; ps += e;
    }
    red[t] = ps; __syncthreads();
    for (int off = 64; off > 0; off >>= 1) {
        if (t < off) red[t] += red[t + off];
        __syncthreads();
    }
    float inv = 1.f / red[0]; __syncthreads();
    #pragma unroll
    for (int q = 0; q < 4; ++q) {
        int j = q * 128 + t;
        float p = sc[j] * inv;
        sc[j] = p;
        atomicAdd(&attnavg[((size_t)(b * SS + i)) * SS + j], p * (1.0f / HH));
    }
    __syncthreads();
    // output: x_attn[b,i,h*64+c] = sum_j p[j]*V[b,j,h*64+c]
    int c = t & 63, half = t >> 6;
    float part = 0.f;
    for (int j = half * 256; j < half * 256 + 256; ++j)
        part += sc[j] * Vf[((size_t)(b * SS + j)) * DD + h * 64 + c];
    red[t] = part; __syncthreads();
    if (t < 64)
        xattn[((size_t)(b * SS + i)) * DD + h * 64 + t] = red[t] + red[t + 64];
}

// ---------------------------------------------------------------------------
// Fused gauge transport: per block (b, i-tile of 16), loop j:
//   lie_i tile [16 x 64] via WMMA over d=512 (x tile stationary in LDS),
//   M = lie_j - lie_i, Gamma = Taylor-6 exp(M) per-row-in-registers,
//   u_t[i,m,:] += attn_avg[i,j] * Gamma[m,:] @ u_chunks[j].
// 128 threads = 4 waves; wave w owns mn-columns [w*16, w*16+16).
// Thread t owns (matrix mi = t>>3, row mm = t&7) in the exp/accum phase.
// ---------------------------------------------------------------------------
__global__ __launch_bounds__(128) void gamma_fiber_kernel(
    const float* __restrict__ x, const __bf16* __restrict__ Abf,
    const float* __restrict__ liej, const float* __restrict__ attnavg,
    const float* __restrict__ uch, float* __restrict__ ut)
{
    const int blk = blockIdx.x;
    const int b = blk >> 5;          // S/16 == 32 tiles per batch
    const int i0 = (blk & 31) * 16;
    const int t = threadIdx.x;
    const int wave = t >> 5, lane = t & 31;
    const int lLo = lane & 15, lHi = lane >> 4;
    const int mi = t >> 3, mm = t & 7;

    __shared__ __align__(32) __bf16 xld[16 * 512];   // 16 KB  (i-tile, bf16)
    __shared__ __align__(32) __bf16 Ald[64 * 128];   // 16 KB  (A chunk [mn][d])
    __shared__ float lield[16 * 64];                 // 4 KB   (-lie tiles)
    __shared__ float ucld[512];                      // u_chunks[j] (8x64)
    __shared__ float ljld[64];
    __shared__ float wld[16];

    // stage x i-tile as bf16
    {
        int row = t >> 3, base = (t & 7) * 64;
        const float* src = x + ((size_t)(b * SS + i0 + row)) * DD + base;
        for (int e = 0; e < 64; ++e)
            xld[row * 512 + base + e] = (__bf16)src[e];
    }

    float acc[64];
    for (int c = 0; c < 64; ++c) acc[c] = 0.f;

    for (int j = 0; j < SS; ++j) {
        __syncthreads();   // protect lield/ucld/ljld/wld from previous iter
        if (t < 64) ljld[t] = liej[((size_t)(b * SS + j)) * 64 + t];
        if (t < 16) wld[t] = attnavg[((size_t)(b * SS + i0 + t)) * SS + j];
        {
            const float* src = uch + ((size_t)(b * SS + j)) * DD;
            #pragma unroll
            for (int q = 0; q < 4; ++q) ucld[t * 4 + q] = src[t * 4 + q];
        }

        v8f cacc = (v8f){};
        for (int dc = 0; dc < 4; ++dc) {          // d chunks of 128
            __syncthreads();
            {   // stage A[b,j, dc*128 + t, 0..63] -> Ald[mn][t]
                const __bf16* src =
                    Abf + (((size_t)(b * SS + j)) * DD + dc * 128 + t) * 64;
                for (int mn = 0; mn < 64; ++mn)
                    Ald[mn * 128 + t] = src[mn];
            }
            __syncthreads();
            #pragma unroll
            for (int ks = 0; ks < 4; ++ks) {      // K steps of 32
                int dbase = dc * 128 + ks * 32;
                union { v16bf v; v8bf h[2]; } ua;
                ua.h[0] = *(const v8bf*)&xld[lLo * 512 + dbase + lHi * 16];
                ua.h[1] = *(const v8bf*)&xld[lLo * 512 + dbase + lHi * 16 + 8];
                v16bf bm = *(const v16bf*)
                    &Ald[(wave * 16 + lLo) * 128 + ks * 32 + lHi * 16];
                cacc = __builtin_amdgcn_wmma_f32_16x16x32_bf16(
                    false, ua.v, false, bm, (short)0, cacc, false, false);
            }
        }
        // M = lie_j - lie_i  ( = -lie )
        #pragma unroll
        for (int r = 0; r < 8; ++r) {
            int ii = lHi * 8 + r;
            int mn = wave * 16 + lLo;
            lield[ii * 64 + mn] = ljld[mn] - cacc[r];
        }
        __syncthreads();

        // Gamma row (mi, mm) via Taylor: G = I + sum M^n/n!
        float rr[8], G[8];
        #pragma unroll
        for (int n = 0; n < 8; ++n) { rr[n] = (n == mm) ? 1.f : 0.f; G[n] = rr[n]; }
        for (int term = 1; term <= 6; ++term) {
            float invn = 1.0f / (float)term;
            float nr[8];
            #pragma unroll
            for (int nn = 0; nn < 8; ++nn) {
                float s = 0.f;
                #pragma unroll
                for (int p = 0; p < 8; ++p)
                    s += rr[p] * lield[mi * 64 + p * 8 + nn];
                nr[nn] = s * invn;
            }
            #pragma unroll
            for (int nn = 0; nn < 8; ++nn) { rr[nn] = nr[nn]; G[nn] += nr[nn]; }
        }
        // u_t[mi, mm, c] += w * sum_n G[n] * uc[n, c]
        float wv = wld[mi];
        for (int c = 0; c < 64; ++c) {
            float s = 0.f;
            #pragma unroll
            for (int n = 0; n < 8; ++n) s += G[n] * ucld[n * 64 + c];
            acc[c] += wv * s;
        }
    }

    float* dst = ut + ((size_t)(b * SS + i0 + mi)) * DD + mm * 64;
    for (int c = 0; c < 64; ++c) dst[c] = acc[c];
}

// ---------------------------------------------------------------------------
extern "C" void kernel_launch(void* const* d_in, const int* in_sizes, int n_in,
                              void* d_out, int out_size, void* d_ws, size_t ws_size,
                              hipStream_t stream)
{
    (void)in_sizes; (void)n_in; (void)out_size; (void)ws_size;
    const float* x    = (const float*)d_in[0];
    const float* u    = (const float*)d_in[1];
    const float* W1   = (const float*)d_in[2];
    const float* b1   = (const float*)d_in[3];
    const float* ln_g = (const float*)d_in[4];
    const float* ln_b = (const float*)d_in[5];
    const float* W2   = (const float*)d_in[6];
    const float* b2   = (const float*)d_in[7];
    const float* Wq   = (const float*)d_in[8];
    const float* bq   = (const float*)d_in[9];
    const float* Wk   = (const float*)d_in[10];
    const float* bk   = (const float*)d_in[11];
    const float* Wv   = (const float*)d_in[12];
    const float* bv   = (const float*)d_in[13];
    const float* Wo   = (const float*)d_in[14];
    const float* bo   = (const float*)d_in[15];
    const float* Wfp  = (const float*)d_in[16];
    const float* bfp  = (const float*)d_in[17];
    const float* Wfo  = (const float*)d_in[18];
    const float* bfo  = (const float*)d_in[19];

    const int M = BB * SS;            // 1024
    float* x_out = (float*)d_out;                 // [1024, 512]
    float* u_out = (float*)d_out + M * DD;        // [1024, 64]

    // workspace carve (all chunks 256B-multiple sized)
    char* p = (char*)d_ws;
    float*  hpre    = (float*)p;  p += (size_t)M * 2048 * 4;     // 8 MB
    float*  hln     = (float*)p;  p += (size_t)M * 2048 * 4;     // 8 MB
    __bf16* Abf     = (__bf16*)p; p += (size_t)M * 32768 * 2;    // 64 MB
    float*  Qf      = (float*)p;  p += (size_t)M * DD * 4;
    float*  Kf      = (float*)p;  p += (size_t)M * DD * 4;
    float*  Vf      = (float*)p;  p += (size_t)M * DD * 4;
    float*  xattn   = (float*)p;  p += (size_t)M * DD * 4;
    float*  attnavg = (float*)p;  p += (size_t)BB * SS * SS * 4; // 2 MB
    float*  liej    = (float*)p;  p += (size_t)M * 64 * 4;
    float*  uch     = (float*)p;  p += (size_t)M * DD * 4;
    float*  ut      = (float*)p;  p += (size_t)M * DD * 4;

    // 0. zero attn_avg accumulator (atomics target; deterministic per call)
    {
        int n = BB * SS * SS;
        zero_kernel<<<(n + 255) / 256, 256, 0, stream>>>(attnavg, n);
    }
    // 1. hpre = gelu(x@W1 + b1)
    wmma_gemm_kernel<<<dim3(2048 / 64, M / 64), 128, 0, stream>>>(
        x, W1, b1, nullptr, hpre, nullptr, M, 2048, DD, 1);
    // 2. LayerNorm
    ln_kernel<<<M, 256, 0, stream>>>(hpre, ln_g, ln_b, hln, 2048);
    // 3. A = bf16(hln@W2 + b2)   -- dominant 137 GFLOP GEMM
    wmma_gemm_kernel<<<dim3(32768 / 64, M / 64), 128, 0, stream>>>(
        hln, W2, b2, nullptr, nullptr, Abf, M, 32768, 2048, 0);
    // 4. Q, K, V projections
    wmma_gemm_kernel<<<dim3(DD / 64, M / 64), 128, 0, stream>>>(
        x, Wq, bq, nullptr, Qf, nullptr, M, DD, DD, 0);
    wmma_gemm_kernel<<<dim3(DD / 64, M / 64), 128, 0, stream>>>(
        x, Wk, bk, nullptr, Kf, nullptr, M, DD, DD, 0);
    wmma_gemm_kernel<<<dim3(DD / 64, M / 64), 128, 0, stream>>>(
        x, Wv, bv, nullptr, Vf, nullptr, M, DD, DD, 0);
    // 5. softmax attention
    attn_kernel<<<dim3(SS, HH, BB), 128, 0, stream>>>(Qf, Kf, Vf, attnavg, xattn);
    // 6. x_out = x + xattn@Wo + bo
    wmma_gemm_kernel<<<dim3(DD / 64, M / 64), 128, 0, stream>>>(
        xattn, Wo, bo, x, x_out, nullptr, M, DD, DD, 0);
    // 7. lie_j
    liej_kernel<<<M, 64, 0, stream>>>(x, Abf, liej);
    // 8. u_chunks = u@Wfp + bfp
    wmma_gemm_kernel<<<dim3(DD / 64, M / 64), 128, 0, stream>>>(
        u, Wfp, bfp, nullptr, uch, nullptr, M, DD, 64, 0);
    // 9. fused Gamma + fiber transport
    gamma_fiber_kernel<<<BB * (SS / 16), 128, 0, stream>>>(
        x, Abf, liej, attnavg, uch, ut);
    // 10. u_out = u + u_t@Wfo + bfo
    wmma_gemm_kernel<<<dim3(1, M / 64), 128, 0, stream>>>(
        ut, Wfo, bfo, u, u_out, nullptr, M, 64, DD, 0);
}
